// RNN_language_model_74242804679360
// MI455X (gfx1250) — matrix operand order repaired and checked by reading, not compiled, not run on hardware
//
#include <hip/hip_runtime.h>
#include <hip/hip_bf16.h>
#include <math.h>

// ---------------------------------------------------------------------------
// Types for CDNA5 WMMA (wave32): v_wmma_f32_16x16x32_bf16
// ---------------------------------------------------------------------------
typedef __attribute__((ext_vector_type(16))) __bf16 v16bf;
typedef __attribute__((ext_vector_type(8)))  __bf16 v8bf;
typedef __attribute__((ext_vector_type(8)))  float  v8f;

#define B_SZ   64
#define T_SZ   101
#define TS     100          // T-1 timesteps
#define V_SZ   8000
#define H_SZ   512
#define G_SZ   2048         // 4*H
#define KCAT   1024         // H (x) + H (h)
#define ROWS   (TS * B_SZ)  // 6400
#define EPSBN  1e-5f

__device__ __forceinline__ float sigm(float x) { return 1.0f / (1.0f + __expf(-x)); }

// Async copy of one 16-byte chunk global -> LDS (per-lane addresses, ASYNCcnt).
__device__ __forceinline__ void async_copy_b128(unsigned lds_off, const void* gptr) {
  unsigned long long ga = (unsigned long long)gptr;
  asm volatile("global_load_async_to_lds_b128 %0, %1, off"
               :: "v"(lds_off), "v"(ga) : "memory");
}
__device__ __forceinline__ void async_wait0() {
  asm volatile("s_wait_asynccnt 0x0" ::: "memory");
}

// ---------------------------------------------------------------------------
// WMMA fragment layouts, per ISA 7.12.2 (wave32):
//  A (16x32 bf16): lane[3:0]=row; lane[4]=0 -> K = kb+0..7, kb+16..23
//                                 lane[4]=1 -> K = kb+8..15, kb+24..31
//  B (32x16 bf16): lane[3:0]=col; same K split as A (kb+0..7,16..23 / kb+8..15,24..31)
//  C/D (16x16 f32): lane[3:0]=col; vgpr e = row m0+e (lane<16) / m0+8+e
// ---------------------------------------------------------------------------
__device__ __forceinline__ v16bf load_fragB(const __bf16* __restrict__ W, int ldw,
                                            int n0, int kb, int lane) {
  int col = n0 + (lane & 15);
  int k0  = kb + ((lane & 16) ? 8 : 0);
  const __bf16* p = W + (size_t)col * ldw + k0;
  v8bf lo = *(const v8bf*)(p);
  v8bf hi = *(const v8bf*)(p + 16);
  v16bf b;
#pragma unroll
  for (int i = 0; i < 8; ++i) { b[i] = lo[i]; b[8 + i] = hi[i]; }
  return b;
}

// A fragment straight from an LDS tile sA[16][ldk] (kept in AS(3) by macro use).
#define LOAD_FRAGA_LDS(frag, sA, ldk, kb, lane)                                \
  do {                                                                         \
    int _row = (lane) & 15;                                                    \
    int _k0  = (kb) + (((lane) & 16) ? 8 : 0);                                 \
    v8bf _lo = *(const v8bf*)&(sA)[_row * (ldk) + _k0];                        \
    v8bf _hi = *(const v8bf*)&(sA)[_row * (ldk) + _k0 + 16];                   \
    _Pragma("unroll")                                                          \
    for (int _i = 0; _i < 8; ++_i) { (frag)[_i] = _lo[_i]; (frag)[8+_i] = _hi[_i]; } \
  } while (0)

// ---------------------------------------------------------------------------
// Prep kernels
// ---------------------------------------------------------------------------
__global__ void build_wcat_kernel(const float* __restrict__ Wih,
                                  const float* __restrict__ Whh,
                                  __bf16* __restrict__ out) {   // [2048,1024]
  int idx = blockIdx.x * blockDim.x + threadIdx.x;
  if (idx >= G_SZ * KCAT) return;
  int n = idx >> 10, k = idx & 1023;
  float v = (k < H_SZ) ? Wih[(size_t)n * H_SZ + k] : Whh[(size_t)n * H_SZ + (k - H_SZ)];
  out[idx] = (__bf16)v;
}

__global__ void cvt_bf16_kernel(const float* __restrict__ src,
                                __bf16* __restrict__ dst, int n) {
  int i = blockIdx.x * blockDim.x + threadIdx.x;
  if (i < n) dst[i] = (__bf16)src[i];
}

__global__ void embed_gather_kernel(const int* __restrict__ x,     // [64,101]
                                    const float* __restrict__ emb, // [V,512]
                                    __bf16* __restrict__ xs) {     // [6400,512] row = t*64+b
  int r = blockIdx.x;
  int t = r >> 6, b = r & 63;
  int tok = x[b * T_SZ + t];
  const float* e = emb + (size_t)tok * H_SZ;
  __bf16* o = xs + (size_t)r * H_SZ;
  for (int j = threadIdx.x; j < H_SZ; j += blockDim.x) o[j] = (__bf16)e[j];
}

__global__ void init_state_kernel(float* c1, float* c2, float* c3,
                                  __bf16* h1, __bf16* h2, __bf16* h3,
                                  float* loss_slot) {
  int i = blockIdx.x * blockDim.x + threadIdx.x;
  if (i < B_SZ * H_SZ) {
    c1[i] = 0.f; c2[i] = 0.f; c3[i] = 0.f;
    __bf16 z = (__bf16)0.0f;
    h1[i] = z; h2[i] = z; h3[i] = z;
  }
  if (i == 0) *loss_slot = 0.f;
}

// ---------------------------------------------------------------------------
// Per-step gates GEMM: gates[64,2048] = [x_t | h] @ Wcat^T + bias
// grid(4 ngroups, 4 mtiles) x 256 threads. Block stages A tile (16x1024 bf16,
// 32KB) into LDS with async-to-LDS; each wave computes 4 N-tiles. All 4 B
// fragments are loaded up front so WMMAs drain the load queue incrementally.
// ---------------------------------------------------------------------------
__global__ void gates_gemm_kernel(const __bf16* __restrict__ xpart, // [64,512]
                                  const __bf16* __restrict__ hpart, // [64,512]
                                  const __bf16* __restrict__ Wcat,  // [2048,1024]
                                  const float*  __restrict__ bias,  // [2048]
                                  float* __restrict__ gates) {      // [64,2048]
  __shared__ __bf16 sA[16 * KCAT];                 // 32 KB
  int wave = threadIdx.x >> 5;
  int lane = threadIdx.x & 31;
  int m0 = blockIdx.y * 16;

  // Stage A: 16 rows x 1024 bf16 = 2048 chunks of 16B; 256 threads x 8 iters.
#pragma unroll
  for (int it = 0; it < 8; ++it) {
    int i = it * 256 + threadIdx.x;                // chunk id
    int row = i >> 7;                              // 128 chunks per row
    int k   = (i & 127) * 8;
    const __bf16* src = (k < H_SZ) ? (xpart + (size_t)(m0 + row) * H_SZ + k)
                                   : (hpart + (size_t)(m0 + row) * H_SZ + (k - H_SZ));
    async_copy_b128((unsigned)(size_t)&sA[row * KCAT + k], src);
  }
  async_wait0();
  __syncthreads();

  int g  = blockIdx.x * 8 + wave;                  // 0..31 n-group
  int n0 = g * 64;
  v8f acc[4];
#pragma unroll
  for (int j = 0; j < 4; ++j) {
    float bv = bias[n0 + 16 * j + (lane & 15)];
#pragma unroll
    for (int i = 0; i < 8; ++i) acc[j][i] = bv;
  }
#pragma unroll 4
  for (int kb = 0; kb < KCAT; kb += 32) {
    v16bf a;
    LOAD_FRAGA_LDS(a, sA, KCAT, kb, lane);
    __builtin_prefetch(Wcat + (size_t)(n0 + (lane & 15)) * KCAT + kb + 128, 0, 1);
    v16bf b[4];
#pragma unroll
    for (int j = 0; j < 4; ++j) b[j] = load_fragB(Wcat, KCAT, n0 + 16 * j, kb, lane);
#pragma unroll
    for (int j = 0; j < 4; ++j)
      acc[j] = __builtin_amdgcn_wmma_f32_16x16x32_bf16(false, a, false, b[j],
                                                       (short)0, acc[j], false, false);
  }
  int mb = m0 + ((lane & 16) ? 8 : 0);
#pragma unroll
  for (int j = 0; j < 4; ++j) {
    int n = n0 + 16 * j + (lane & 15);
#pragma unroll
    for (int e = 0; e < 8; ++e) gates[(size_t)(mb + e) * G_SZ + n] = acc[j][e];
  }
}

// ---------------------------------------------------------------------------
// LSTM cell + BatchNorm(train) + locked dropout. One block, 512 threads;
// thread j owns channel j -> batch stats are thread-local registers.
// ---------------------------------------------------------------------------
__global__ void lstm_cell_bn_kernel(const float* __restrict__ gates, // [64,2048]
                                    float*  __restrict__ c,          // [64,512]
                                    __bf16* __restrict__ h_bf,       // [64,512]
                                    __bf16* __restrict__ bnout,      // [64,512]
                                    const float* __restrict__ gamma,
                                    const float* __restrict__ beta,
                                    const float* __restrict__ mask) {
  int j = threadIdx.x;
  float hloc[B_SZ];
  float s = 0.f, ss = 0.f;
#pragma unroll
  for (int b = 0; b < B_SZ; ++b) {
    const float* g = gates + (size_t)b * G_SZ;
    float ig = sigm(g[j]);
    float fg = sigm(g[H_SZ + j]);
    float gg = tanhf(g[2 * H_SZ + j]);
    float og = sigm(g[3 * H_SZ + j]);
    float cn = fg * c[b * H_SZ + j] + ig * gg;
    c[b * H_SZ + j] = cn;
    float h = og * tanhf(cn);
    h_bf[b * H_SZ + j] = (__bf16)h;
    hloc[b] = h;
    s += h; ss += h * h;
  }
  float mu   = s * (1.0f / B_SZ);
  float var  = ss * (1.0f / B_SZ) - mu * mu;
  float rstd = rsqrtf(var + EPSBN);
  float ga = gamma[j], be = beta[j];
#pragma unroll
  for (int b = 0; b < B_SZ; ++b) {
    float bn = ga * (hloc[b] - mu) * rstd + be;
    bnout[b * H_SZ + j] = (__bf16)(bn * mask[b * H_SZ + j]);
  }
}

// ---------------------------------------------------------------------------
// Decoder GEMM: logits[6400,8000] = Hout @ Wd^T + bd, stored transposed into
// d_out as outputs[b][v][t] at offset 1. grid(16 ngroups, 400 mtiles) x 256.
// Block stages A (16x512 bf16, 16KB) via async-to-LDS; wave = 4 N-tiles with
// all 4 B fragments loaded up front per k-block.
// ---------------------------------------------------------------------------
__global__ void decoder_gemm_kernel(const __bf16* __restrict__ Hout, // [6400,512]
                                    const __bf16* __restrict__ Wd,   // [8000,512]
                                    const float*  __restrict__ bd,   // [8000]
                                    float* __restrict__ out) {       // d_out base
  __shared__ __bf16 sA[16 * H_SZ];                 // 16 KB
  int wave = threadIdx.x >> 5;
  int lane = threadIdx.x & 31;
  int m0 = blockIdx.y * 16;

  // Stage A: 16 rows x 512 bf16 = 1024 chunks of 16B; 256 threads x 4 iters.
#pragma unroll
  for (int it = 0; it < 4; ++it) {
    int i = it * 256 + threadIdx.x;
    int row = i >> 6;                              // 64 chunks per row
    int k   = (i & 63) * 8;
    async_copy_b128((unsigned)(size_t)&sA[row * H_SZ + k],
                    Hout + (size_t)(m0 + row) * H_SZ + k);
  }
  async_wait0();
  __syncthreads();

  int g = blockIdx.x * 8 + wave;                   // 0..127
  if (g < 125) {
    int n0 = g * 64;
    v8f acc[4];
#pragma unroll
    for (int j = 0; j < 4; ++j) {
      float bv = bd[n0 + 16 * j + (lane & 15)];
#pragma unroll
      for (int i = 0; i < 8; ++i) acc[j][i] = bv;
    }
#pragma unroll 4
    for (int kb = 0; kb < H_SZ; kb += 32) {
      v16bf a;
      LOAD_FRAGA_LDS(a, sA, H_SZ, kb, lane);
      __builtin_prefetch(Wd + (size_t)(n0 + (lane & 15)) * H_SZ + kb + 128, 0, 1);
      v16bf b[4];
#pragma unroll
      for (int j = 0; j < 4; ++j) b[j] = load_fragB(Wd, H_SZ, n0 + 16 * j, kb, lane);
#pragma unroll
      for (int j = 0; j < 4; ++j)
        acc[j] = __builtin_amdgcn_wmma_f32_16x16x32_bf16(false, a, false, b[j],
                                                         (short)0, acc[j], false, false);
    }
    int mb = m0 + ((lane & 16) ? 8 : 0);
#pragma unroll
    for (int j = 0; j < 4; ++j) {
      int n = n0 + 16 * j + (lane & 15);
#pragma unroll
      for (int e = 0; e < 8; ++e) {
        int m = mb + e;                            // row = t*64 + b
        int t = m >> 6, b_ = m & 63;
        out[1 + ((size_t)b_ * V_SZ + n) * TS + t] = acc[j][e];
      }
    }
  }
}

// ---------------------------------------------------------------------------
// Loss: one-pass online softmax. Block = batch b; thread t<100 owns row (b,t);
// reads outputs[b][v][t] (coalesced over t) for v = 0..7999.
// ---------------------------------------------------------------------------
__global__ void loss_kernel(const float* __restrict__ out,
                            const int*   __restrict__ x,
                            float* __restrict__ loss_slot) {
  int b = blockIdx.x;
  int t = threadIdx.x;
  if (t >= TS) return;
  const float* base = out + 1 + (size_t)b * V_SZ * TS + t;
  int target = x[b * T_SZ + t + 1];
  float m = -INFINITY, s = 0.f, tv = 0.f;
  for (int v = 0; v < V_SZ; ++v) {
    float val = base[(size_t)v * TS];
    if (v == target) tv = val;
    float mn = fmaxf(m, val);
    s = s * __expf(m - mn) + __expf(val - mn);
    m = mn;
  }
  float logp = tv - m - __logf(s);
  atomicAdd(loss_slot, -logp * (1.0f / (float)ROWS));
}

// ---------------------------------------------------------------------------
// Host launcher
// ---------------------------------------------------------------------------
extern "C" void kernel_launch(void* const* d_in, const int* in_sizes, int n_in,
                              void* d_out, int out_size, void* d_ws, size_t ws_size,
                              hipStream_t stream) {
  (void)in_sizes; (void)n_in; (void)out_size; (void)ws_size;

  const int*   x   = (const int*)  d_in[0];
  const float* emb = (const float*)d_in[1];
  const float* Wd  = (const float*)d_in[2];
  const float* bd  = (const float*)d_in[3];
  const float *Wih[3], *Whh[3], *bia[3], *gam[3], *bet[3], *msk[3];
  for (int l = 0; l < 3; ++l) {
    Wih[l] = (const float*)d_in[4 + l * 6 + 0];
    Whh[l] = (const float*)d_in[4 + l * 6 + 1];
    bia[l] = (const float*)d_in[4 + l * 6 + 2];
    gam[l] = (const float*)d_in[4 + l * 6 + 3];
    bet[l] = (const float*)d_in[4 + l * 6 + 4];
    msk[l] = (const float*)d_in[4 + l * 6 + 5];
  }
  float* out = (float*)d_out;

  // Workspace carve-up (256B aligned)
  char* w = (char*)d_ws;
  auto take = [&](size_t bytes) { char* p = w; w += (bytes + 255) & ~(size_t)255; return p; };
  __bf16* Wcat[3]; float* G[3]; float* C[3]; __bf16* Hbf[3];
  for (int l = 0; l < 3; ++l) Wcat[l] = (__bf16*)take((size_t)G_SZ * KCAT * 2);
  __bf16* WdBf   = (__bf16*)take((size_t)V_SZ * H_SZ * 2);
  __bf16* XsBf   = (__bf16*)take((size_t)ROWS * H_SZ * 2);
  __bf16* HoutBf = (__bf16*)take((size_t)ROWS * H_SZ * 2);
  for (int l = 0; l < 3; ++l) G[l]   = (float*)take((size_t)B_SZ * G_SZ * 4);
  for (int l = 0; l < 3; ++l) C[l]   = (float*)take((size_t)B_SZ * H_SZ * 4);
  for (int l = 0; l < 3; ++l) Hbf[l] = (__bf16*)take((size_t)B_SZ * H_SZ * 2);
  __bf16* BN1 = (__bf16*)take((size_t)B_SZ * H_SZ * 2);
  __bf16* BN2 = (__bf16*)take((size_t)B_SZ * H_SZ * 2);

  // ---- Prep ----
  for (int l = 0; l < 3; ++l)
    build_wcat_kernel<<<(G_SZ * KCAT + 255) / 256, 256, 0, stream>>>(Wih[l], Whh[l], Wcat[l]);
  cvt_bf16_kernel<<<(V_SZ * H_SZ + 255) / 256, 256, 0, stream>>>(Wd, WdBf, V_SZ * H_SZ);
  embed_gather_kernel<<<ROWS, 256, 0, stream>>>(x, emb, XsBf);
  init_state_kernel<<<(B_SZ * H_SZ + 255) / 256, 256, 0, stream>>>(
      C[0], C[1], C[2], Hbf[0], Hbf[1], Hbf[2], out);

  // ---- Recurrence: 100 steps x 3 layers ----
  dim3 ggrid(4, 4);
  for (int t = 0; t < TS; ++t) {
    const __bf16* xt = XsBf + (size_t)t * B_SZ * H_SZ;
    gates_gemm_kernel<<<ggrid, 256, 0, stream>>>(xt,  Hbf[0], Wcat[0], bia[0], G[0]);
    lstm_cell_bn_kernel<<<1, H_SZ, 0, stream>>>(G[0], C[0], Hbf[0], BN1,
                                                gam[0], bet[0], msk[0]);
    gates_gemm_kernel<<<ggrid, 256, 0, stream>>>(BN1, Hbf[1], Wcat[1], bia[1], G[1]);
    lstm_cell_bn_kernel<<<1, H_SZ, 0, stream>>>(G[1], C[1], Hbf[1], BN2,
                                                gam[1], bet[1], msk[1]);
    gates_gemm_kernel<<<ggrid, 256, 0, stream>>>(BN2, Hbf[2], Wcat[2], bia[2], G[2]);
    lstm_cell_bn_kernel<<<1, H_SZ, 0, stream>>>(G[2], C[2], Hbf[2],
                                                HoutBf + (size_t)t * B_SZ * H_SZ,
                                                gam[2], bet[2], msk[2]);
  }

  // ---- Decoder + loss ----
  dim3 dgrid(16, 400);
  decoder_gemm_kernel<<<dgrid, 256, 0, stream>>>(HoutBf, WdBf, bd, out);
  loss_kernel<<<B_SZ, 128, 0, stream>>>(out, x, out);
}
